// LightGlueMatchAssignmentLayer_39281770889673
// MI455X (gfx1250) — matrix operand order repaired and checked by reading, not compiled, not run on hardware
//
#include <hip/hip_runtime.h>
#include <hip/hip_bf16.h>
#include <cfloat>

// ---------------------------------------------------------------------------
// LightGlue match-assignment layer for MI455X (gfx1250, wave32, WMMA).
// bf16x3 (hi/lo split) GEMMs on the bf16 WMMA pipe; split materialized ONCE
// so the similarity inner loop is pure loads + v_wmma. Row/col logsumexp
// partials are fused into the GEMM epilogue to cut two full 134MB passes.
// ---------------------------------------------------------------------------

typedef __attribute__((ext_vector_type(16))) __bf16 v16bf;
typedef __attribute__((ext_vector_type(8)))  __bf16 v8bf;
typedef __attribute__((ext_vector_type(8)))  float  v8f;

#define DD   256
#define NN   2048
#define NP1  2049
static constexpr long PAIR_STRIDE = (long)NP1 * NP1;   // 4198401 floats per pair

// ---- workspace layout (bytes) ----
static constexpr size_t SZ_MD  = (size_t)32768 * 256 * 2;   // one bf16 mdesc plane
static constexpr size_t SZ_W   = (size_t)256 * 256 * 2;     // one bf16 weight plane
static constexpr size_t SZ_RP  = (size_t)8 * 32 * 2048 * 4; // row partials (m or s)
static constexpr size_t SZ_CP  = (size_t)8 * 128 * 2048 * 4;// col partials (m or s)
static constexpr size_t SZ_LSE = (size_t)8 * 2048 * 4;
static constexpr size_t SZ_LS  = (size_t)32768 * 4;

static constexpr size_t OFF_MDH  = 0;
static constexpr size_t OFF_MDL  = OFF_MDH + SZ_MD;
static constexpr size_t OFF_WH   = OFF_MDL + SZ_MD;
static constexpr size_t OFF_WL   = OFF_WH  + SZ_W;
static constexpr size_t OFF_RPM  = OFF_WL  + SZ_W;
static constexpr size_t OFF_RPS  = OFF_RPM + SZ_RP;
static constexpr size_t OFF_CPM  = OFF_RPS + SZ_RP;
static constexpr size_t OFF_CPS  = OFF_CPM + SZ_CP;
static constexpr size_t OFF_LROW = OFF_CPS + SZ_CP;
static constexpr size_t OFF_LCOL = OFF_LROW + SZ_LSE;
static constexpr size_t OFF_LSP  = OFF_LCOL + SZ_LSE;
static constexpr size_t OFF_LSN  = OFF_LSP  + SZ_LS;
// total ~55 MB of ws

__device__ __forceinline__ float4 ldg4(const float* p) {
  return *reinterpret_cast<const float4*>(p);
}
__device__ __forceinline__ v8bf ldg8bf(const __bf16* p) {
  return *reinterpret_cast<const v8bf*>(p);
}
__device__ __forceinline__ v16bf cat8(v8bf a, v8bf b) {
  return __builtin_shufflevector(a, b, 0,1,2,3,4,5,6,7,8,9,10,11,12,13,14,15);
}

// split 16 fp32 values into bf16 hi/lo vectors (x ~= hi + lo)
__device__ __forceinline__ void split16(const float* f, v16bf& hi, v16bf& lo) {
#pragma unroll
  for (int i = 0; i < 16; ++i) {
    float x = f[i];
    __bf16 h = (__bf16)x;
    hi[i] = h;
    lo[i] = (__bf16)(x - (float)h);
  }
}

// bf16x3: C += Ahi*Bhi + Ahi*Blo + Alo*Bhi  (fp32 accumulate)
__device__ __forceinline__ v8f wmma3(v16bf ah, v16bf al, v16bf bh, v16bf bl, v8f c) {
  c = __builtin_amdgcn_wmma_f32_16x16x32_bf16(false, ah, false, bh, (short)0, c, false, false);
  c = __builtin_amdgcn_wmma_f32_16x16x32_bf16(false, ah, false, bl, (short)0, c, false, false);
  c = __builtin_amdgcn_wmma_f32_16x16x32_bf16(false, al, false, bh, (short)0, c, false, false);
  return c;
}

__device__ __forceinline__ float logsig(float x) {
  return fminf(x, 0.f) - log1pf(expf(-fabsf(x)));
}

// online-LSE combine of (m1,s1) with (m2,s2)
__device__ __forceinline__ void lse_merge(float& m, float& s, float m2, float s2) {
  float mn = fmaxf(m, m2);
  s = s * expf(m - mn) + s2 * expf(m2 - mn);
  m = mn;
}

// -------- pre-split proj_w into bf16 hi/lo planes (tiny, one-shot) --------
__global__ void split_w_kernel(const float* __restrict__ w, __bf16* __restrict__ wh,
                               __bf16* __restrict__ wl) {
  int g = blockIdx.x * 256 + threadIdx.x;   // 16384 threads, 4 elems each
#pragma unroll
  for (int e = 0; e < 4; ++e) {
    int idx = g * 4 + e;
    float x = w[idx];
    __bf16 h = (__bf16)x;
    wh[idx] = h;
    wl[idx] = (__bf16)(x - (float)h);
  }
}

// -------- pass 1: m_desc = (desc @ W^T + b) * 0.25, stored as bf16 hi/lo ----
__global__ void proj_kernel(const float* __restrict__ desc, const __bf16* __restrict__ wh,
                            const __bf16* __restrict__ wl, const float* __restrict__ bias,
                            __bf16* __restrict__ mdh, __bf16* __restrict__ mdl) {
  int tid  = threadIdx.x;
  int wave = tid >> 5, lane = tid & 31;
  int kh   = lane >> 4, l15 = lane & 15;
  int gid  = blockIdx.x * 8 + wave;               // 0..32767 tiles
  int row_tile = gid >> 4;                        // 2048 row tiles
  int col_tile = gid & 15;                        // 16 col tiles
  const float*  arow = desc + (size_t)(row_tile * 16 + l15) * DD;
  const __bf16* bhrow = wh + (size_t)(col_tile * 16 + l15) * DD;
  const __bf16* blrow = wl + (size_t)(col_tile * 16 + l15) * DD;

  v8f acc;
#pragma unroll
  for (int e = 0; e < 8; ++e) acc[e] = 0.f;

  for (int k0 = 0; k0 < DD; k0 += 32) {
    __align__(16) float af[16];
    // A 16x32 bf16 layout: lane holds K in [kh*8, kh*8+7] and [16+kh*8, 16+kh*8+7]
    const float* a = arow + k0 + kh * 8;
    *(float4*)&af[0]  = ldg4(a);
    *(float4*)&af[4]  = ldg4(a + 4);
    *(float4*)&af[8]  = ldg4(a + 16);
    *(float4*)&af[12] = ldg4(a + 20);
    v16bf ah, al;
    split16(af, ah, al);
    // B 32x16 bf16 layout: lane holds K in [kh*8, kh*8+7] and [16+kh*8, 16+kh*8+7], N = l15
    const __bf16* bh8 = bhrow + k0 + kh * 8;
    const __bf16* bl8 = blrow + k0 + kh * 8;
    v16bf bh = cat8(ldg8bf(bh8), ldg8bf(bh8 + 16));
    v16bf bl = cat8(ldg8bf(bl8), ldg8bf(bl8 + 16));
    acc = wmma3(ah, al, bh, bl, acc);
  }

  // C layout: vgpr vr -> M = vr + 8*kh, N = l15; store split bf16 result
  int o = col_tile * 16 + l15;
  float bv = bias[o];
#pragma unroll
  for (int vr = 0; vr < 8; ++vr) {
    int r = row_tile * 16 + kh * 8 + vr;
    float v = (acc[vr] + bv) * 0.25f;
    __bf16 h = (__bf16)v;
    size_t idx = (size_t)r * DD + o;
    mdh[idx] = h;
    mdl[idx] = (__bf16)(v - (float)h);
  }
}

// -------- matchability: m = desc . match_w + b; store logsig(+-m) --------
__global__ void match_kernel(const float* __restrict__ desc, const float* __restrict__ mw,
                             const float* __restrict__ mb, float* __restrict__ lspos,
                             float* __restrict__ lsneg) {
  int tid  = threadIdx.x;
  int row  = blockIdx.x * 8 + (tid >> 5);
  int lane = tid & 31;
  const float* d = desc + (size_t)row * DD + lane * 8;
  const float* w = mw + lane * 8;
  float4 x0 = ldg4(d), x1 = ldg4(d + 4);
  float4 w0 = ldg4(w), w1 = ldg4(w + 4);
  float s = x0.x * w0.x + x0.y * w0.y + x0.z * w0.z + x0.w * w0.w
          + x1.x * w1.x + x1.y * w1.y + x1.z * w1.z + x1.w * w1.w;
#pragma unroll
  for (int off = 16; off >= 1; off >>= 1) s += __shfl_xor(s, off, 32);
  if (lane == 0) {
    float m = s + mb[0];
    lspos[row] = logsig(m);
    lsneg[row] = logsig(-m);
  }
}

// -------- pass 2: masked similarity + fused row/col LSE partials --------
// Each wave: 16 rows x 64 cols strip. Inner loop is pure bf16 loads + WMMA.
__global__ void sim_kernel(const __bf16* __restrict__ mdh, const __bf16* __restrict__ mdl,
                           const int* __restrict__ mask, float* __restrict__ out,
                           float* __restrict__ rpm, float* __restrict__ rps,
                           float* __restrict__ cpm, float* __restrict__ cps) {
  int tid  = threadIdx.x;
  int wave = tid >> 5, lane = tid & 31;
  int kh   = lane >> 4, l15 = lane & 15;
  int gid  = blockIdx.x * 8 + wave;               // 0..32767
  int p    = gid >> 12;                           // 4096 wave-tiles per pair
  int rem  = gid & 4095;
  int i_tile  = rem >> 5;                         // 0..127
  int j_strip = rem & 31;                         // 0..31 (64 cols each)

  const __bf16* ahrow = mdh + ((size_t)(2 * p) * NN + i_tile * 16 + l15) * DD;
  const __bf16* alrow = mdl + ((size_t)(2 * p) * NN + i_tile * 16 + l15) * DD;
  const __bf16* bhbase = mdh + ((size_t)(2 * p + 1) * NN + j_strip * 64 + l15) * DD;
  const __bf16* blbase = mdl + ((size_t)(2 * p + 1) * NN + j_strip * 64 + l15) * DD;

  v8f acc[4];
#pragma unroll
  for (int s = 0; s < 4; ++s)
#pragma unroll
    for (int e = 0; e < 8; ++e) acc[s][e] = 0.f;

  for (int k0 = 0; k0 < DD; k0 += 32) {
    const __bf16* a_h = ahrow + k0 + kh * 8;
    const __bf16* a_l = alrow + k0 + kh * 8;
    v16bf ah = cat8(ldg8bf(a_h), ldg8bf(a_h + 16));
    v16bf al = cat8(ldg8bf(a_l), ldg8bf(a_l + 16));
#pragma unroll
    for (int s = 0; s < 4; ++s) {
      const __bf16* b_h = bhbase + (size_t)s * 16 * DD + k0 + kh * 8;
      const __bf16* b_l = blbase + (size_t)s * 16 * DD + k0 + kh * 8;
      v16bf bh = cat8(ldg8bf(b_h), ldg8bf(b_h + 16));
      v16bf bl = cat8(ldg8bf(b_l), ldg8bf(b_l + 16));
      acc[s] = wmma3(ah, al, bh, bl, acc[s]);
    }
  }

  // ---- mask + store ----
  const int* m0 = mask + (size_t)(2 * p) * NN;
  const int* m1 = mask + (size_t)(2 * p + 1) * NN;
  long obase = (long)p * PAIR_STRIDE;
  float val[4][8];
#pragma unroll
  for (int s = 0; s < 4; ++s) {
    int j  = j_strip * 64 + s * 16 + l15;
    int mj = m1[j];
#pragma unroll
    for (int vr = 0; vr < 8; ++vr) {
      int i  = i_tile * 16 + kh * 8 + vr;
      int mi = m0[i];
      float v = (mi != 0 && mj != 0) ? acc[s][vr] : -FLT_MAX;
      val[s][vr] = v;
      out[obase + (long)i * NP1 + j] = v;
    }
  }

  // ---- fused per-row LSE partial over this 64-col strip ----
  // Row r = i_tile*16 + kh*8 + vr lives in lanes sharing kh; reduce over the
  // 4 in-lane subtiles then butterfly across the 16 lanes of the half-wave.
#pragma unroll
  for (int vr = 0; vr < 8; ++vr) {
    float m = fmaxf(fmaxf(val[0][vr], val[1][vr]), fmaxf(val[2][vr], val[3][vr]));
    float s = expf(val[0][vr] - m) + expf(val[1][vr] - m)
            + expf(val[2][vr] - m) + expf(val[3][vr] - m);
#pragma unroll
    for (int off = 8; off >= 1; off >>= 1)
      lse_merge(m, s, __shfl_xor(m, off, 32), __shfl_xor(s, off, 32));
    if (l15 == 0) {
      int i = i_tile * 16 + kh * 8 + vr;
      int idx = (p * 32 + j_strip) * NN + i;
      rpm[idx] = m;
      rps[idx] = s;
    }
  }

  // ---- fused per-column LSE partial over this 16-row tile ----
  // Column j lives in lane pair (L, L^16): reduce 8 in-lane rows, then one
  // cross-half merge; kh==0 lanes hold the result.
#pragma unroll
  for (int s = 0; s < 4; ++s) {
    float m = val[s][0];
#pragma unroll
    for (int vr = 1; vr < 8; ++vr) m = fmaxf(m, val[s][vr]);
    float ss = 0.f;
#pragma unroll
    for (int vr = 0; vr < 8; ++vr) ss += expf(val[s][vr] - m);
    lse_merge(m, ss, __shfl_xor(m, 16, 32), __shfl_xor(ss, 16, 32));
    if (kh == 0) {
      int j = j_strip * 64 + s * 16 + l15;
      int idx = (p * 128 + i_tile) * NN + j;
      cpm[idx] = m;
      cps[idx] = ss;
    }
  }
}

// -------- combine row partials (32 strips) --------
__global__ void lse_row_comb(const float* __restrict__ rpm, const float* __restrict__ rps,
                             float* __restrict__ lse_row) {
  int g = blockIdx.x * 256 + threadIdx.x;         // 0..16383
  int p = g >> 11, i = g & 2047;
  float m = -FLT_MAX, s = 0.f;
  for (int strip = 0; strip < 32; ++strip) {
    int idx = (p * 32 + strip) * NN + i;
    lse_merge(m, s, rpm[idx], rps[idx]);
  }
  lse_row[g] = m + logf(s);
}

// -------- combine column partials (128 row tiles) --------
__global__ void lse_col_comb(const float* __restrict__ cpm, const float* __restrict__ cps,
                             float* __restrict__ lse_col) {
  int g = blockIdx.x * 256 + threadIdx.x;         // 0..16383
  int p = g >> 11, j = g & 2047;
  float m = -FLT_MAX, s = 0.f;
  for (int t = 0; t < 128; ++t) {
    int idx = (p * 128 + t) * NN + j;
    lse_merge(m, s, cpm[idx], cps[idx]);
  }
  lse_col[g] = m + logf(s);
}

// -------- pass 3: finalize all (n+1)x(n+1) entries --------
__global__ void finalize_kernel(const float* __restrict__ lse_row, const float* __restrict__ lse_col,
                                const float* __restrict__ lspos, const float* __restrict__ lsneg,
                                float* __restrict__ out) {
  int j = blockIdx.x * 256 + threadIdx.x;
  if (j > 2048) return;
  int i = blockIdx.y;
  int p = blockIdx.z;
  long idx = (long)p * PAIR_STRIDE + (long)i * NP1 + j;
  if (i < NN && j < NN) {
    float sv = out[idx];
    out[idx] = (sv - lse_row[p * NN + i]) + (sv - lse_col[p * NN + j])
             + lspos[(2 * p) * NN + i] + lspos[(2 * p + 1) * NN + j];
  } else if (i < NN) {                 // j == 2048: logsig(-m0)
    out[idx] = lsneg[(2 * p) * NN + i];
  } else if (j < NN) {                 // i == 2048: logsig(-m1)
    out[idx] = lsneg[(2 * p + 1) * NN + j];
  } else {
    out[idx] = 0.f;                    // corner
  }
}

extern "C" void kernel_launch(void* const* d_in, const int* in_sizes, int n_in,
                              void* d_out, int out_size, void* d_ws, size_t ws_size,
                              hipStream_t stream) {
  (void)in_sizes; (void)n_in; (void)out_size; (void)ws_size;
  const float* desc = (const float*)d_in[0];
  const int*   mask = (const int*)d_in[1];
  const float* pw   = (const float*)d_in[2];
  const float* pb   = (const float*)d_in[3];
  const float* mw   = (const float*)d_in[4];
  const float* mb   = (const float*)d_in[5];
  float* out = (float*)d_out;
  char*  ws  = (char*)d_ws;

  __bf16* mdh = (__bf16*)(ws + OFF_MDH);
  __bf16* mdl = (__bf16*)(ws + OFF_MDL);
  __bf16* wh  = (__bf16*)(ws + OFF_WH);
  __bf16* wl  = (__bf16*)(ws + OFF_WL);
  float* rpm  = (float*)(ws + OFF_RPM);
  float* rps  = (float*)(ws + OFF_RPS);
  float* cpm  = (float*)(ws + OFF_CPM);
  float* cps  = (float*)(ws + OFF_CPS);
  float* lrow = (float*)(ws + OFF_LROW);
  float* lcol = (float*)(ws + OFF_LCOL);
  float* lsp  = (float*)(ws + OFF_LSP);
  float* lsn  = (float*)(ws + OFF_LSN);

  split_w_kernel <<<64,   256, 0, stream>>>(pw, wh, wl);
  proj_kernel    <<<4096, 256, 0, stream>>>(desc, wh, wl, pb, mdh, mdl);
  match_kernel   <<<4096, 256, 0, stream>>>(desc, mw, mb, lsp, lsn);
  sim_kernel     <<<4096, 256, 0, stream>>>(mdh, mdl, mask, out, rpm, rps, cpm, cps);
  lse_row_comb   <<<64,   256, 0, stream>>>(rpm, rps, lrow);
  lse_col_comb   <<<64,   256, 0, stream>>>(cpm, cps, lcol);
  finalize_kernel<<<dim3(9, 2049, 8), 256, 0, stream>>>(lrow, lcol, lsp, lsn, out);
}
